// CategoricalCNN_54623394070730
// MI455X (gfx1250) — matrix operand order repaired and checked by reading, not compiled, not run on hardware
//
#include <hip/hip_runtime.h>

#define H   480
#define W   320
#define BS  20
#define PADN 6
#define SF  4
#define WIN 32
#define NH  24
#define NW  16
#define NB  (NH*NW)

typedef __attribute__((ext_vector_type(16))) _Float16 v16h;
typedef __attribute__((ext_vector_type(8)))  _Float16 v8h;
typedef __attribute__((ext_vector_type(8)))  float    v8f;

__device__ __forceinline__ int iclamp(int v, int lo, int hi) {
    return v < lo ? lo : (v > hi ? hi : v);
}

// ---------------------------------------------------------------------------
// Generic direct f32 conv (batch 1, NCHW). Used for the classifier branch.
// ---------------------------------------------------------------------------
__global__ void conv2d_f32(const float* __restrict__ in, const float* __restrict__ w,
                           const float* __restrict__ bias, float* __restrict__ out,
                           int Cin, int Cout, int Hin, int Win, int Hout, int Wout,
                           int K, int stride, int pad, int edgePad, int doRelu)
{
    int idx = blockIdx.x * blockDim.x + threadIdx.x;
    int total = Cout * Hout * Wout;
    if (idx >= total) return;
    int x  = idx % Wout;
    int y  = (idx / Wout) % Hout;
    int co = idx / (Wout * Hout);
    float acc = bias[co];
    for (int ci = 0; ci < Cin; ++ci) {
        const float* inc = in + (size_t)ci * Hin * Win;
        const float* wc  = w + ((size_t)co * Cin + ci) * K * K;
        for (int ky = 0; ky < K; ++ky) {
            for (int kx = 0; kx < K; ++kx) {
                int iy = y * stride + ky - pad;
                int ix = x * stride + kx - pad;
                float v = 0.f;
                if (edgePad) {
                    iy = iclamp(iy, 0, Hin - 1);
                    ix = iclamp(ix, 0, Win - 1);
                    v = inc[iy * Win + ix];
                } else if (iy >= 0 && iy < Hin && ix >= 0 && ix < Win) {
                    v = inc[iy * Win + ix];
                }
                acc += v * wc[ky * K + kx];
            }
        }
    }
    if (doRelu) acc = fmaxf(acc, 0.f);
    out[idx] = acc;
}

// ---------------------------------------------------------------------------
// 2x2 max pool, stride 2, f32 NCHW
// ---------------------------------------------------------------------------
__global__ void maxpool2_f32(const float* __restrict__ in, float* __restrict__ out,
                             int C, int Hin, int Win)
{
    int Ho = Hin >> 1, Wo = Win >> 1;
    int idx = blockIdx.x * blockDim.x + threadIdx.x;
    int total = C * Ho * Wo;
    if (idx >= total) return;
    int x = idx % Wo;
    int y = (idx / Wo) % Ho;
    int c = idx / (Wo * Ho);
    const float* p = in + ((size_t)c * Hin + 2 * y) * Win + 2 * x;
    out[idx] = fmaxf(fmaxf(p[0], p[1]), fmaxf(p[Win], p[Win + 1]));
}

// ---------------------------------------------------------------------------
// Sigmoid over the 384 classifier logits -> ws copy + tail of d_out
// ---------------------------------------------------------------------------
__global__ void sigmoid384(const float* __restrict__ in, float* __restrict__ clsvec,
                           float* __restrict__ outTail)
{
    int i = blockIdx.x * blockDim.x + threadIdx.x;
    if (i < NB) {
        float s = 1.f / (1.f + __expf(-in[i]));
        clsvec[i] = s;
        outTail[i] = s;
    }
}

// ---------------------------------------------------------------------------
// Extract 384 blocks of 3x32x32 from the zero-padded input, f32 NCHW per block
// ---------------------------------------------------------------------------
__global__ void extract_blocks(const float* __restrict__ x, float* __restrict__ blocks)
{
    int idx = blockIdx.x * blockDim.x + threadIdx.x;
    int total = NB * 3 * WIN * WIN;
    if (idx >= total) return;
    int xx = idx % WIN;
    int yy = (idx / WIN) % WIN;
    int c  = (idx / (WIN * WIN)) % 3;
    int b  = idx / (3 * WIN * WIN);
    int bh = b / NW, bw = b % NW;
    int gy = bh * BS - PADN + yy;
    int gx = bw * BS - PADN + xx;
    float v = 0.f;
    if (gy >= 0 && gy < H && gx >= 0 && gx < W)
        v = x[((size_t)c * H + gy) * W + gx];
    blocks[idx] = v;
}

// ---------------------------------------------------------------------------
// Direct conv for Cin=3 per-block layers (lt1, cx1): zero pad 1, K=3, ReLU.
// Writes NHWC f16 activations [NB][1024][Cpad] with zero channel padding.
// ---------------------------------------------------------------------------
__global__ void conv_block_c3(const float* __restrict__ blocks, const float* __restrict__ w,
                              const float* __restrict__ bias, _Float16* __restrict__ out,
                              int Cout, int Cpad, int doRelu)
{
    int idx = blockIdx.x * blockDim.x + threadIdx.x;
    int total = NB * WIN * WIN * Cpad;
    if (idx >= total) return;
    int co = idx % Cpad;
    int p  = (idx / Cpad) % (WIN * WIN);
    int b  = idx / (Cpad * WIN * WIN);
    if (co >= Cout) { out[idx] = (_Float16)0.f; return; }
    int y = p / WIN, x = p % WIN;
    float acc = bias[co];
    for (int ci = 0; ci < 3; ++ci) {
        for (int ky = 0; ky < 3; ++ky) {
            int iy = y + ky - 1;
            if ((unsigned)iy >= (unsigned)WIN) continue;
            for (int kx = 0; kx < 3; ++kx) {
                int ix = x + kx - 1;
                if ((unsigned)ix >= (unsigned)WIN) continue;
                acc += blocks[(((size_t)b * 3 + ci) * WIN + iy) * WIN + ix] *
                       w[((co * 3 + ci) * 3 + ky) * 3 + kx];
            }
        }
    }
    if (doRelu) acc = fmaxf(acc, 0.f);
    out[idx] = (_Float16)acc;
}

// ---------------------------------------------------------------------------
// Weight transpose OIHW f32 -> [tap][cout][cin_pad] f16 (cin-fastest, padded)
// ---------------------------------------------------------------------------
__global__ void prep_wt(const float* __restrict__ w, _Float16* __restrict__ wt,
                        int Cout, int Cin, int Cpad)
{
    int idx = blockIdx.x * blockDim.x + threadIdx.x;
    int total = 9 * Cout * Cpad;
    if (idx >= total) return;
    int ci  = idx % Cpad;
    int co  = (idx / Cpad) % Cout;
    int tap = idx / (Cpad * Cout);
    int ky = tap / 3, kx = tap % 3;
    float v = (ci < Cin) ? w[(((size_t)co * Cin + ci) * 3 + ky) * 3 + kx] : 0.f;
    wt[idx] = (_Float16)v;
}

// ---------------------------------------------------------------------------
// WMMA implicit-GEMM 3x3 conv over 32x32 blocks, zero pad 1.
//   act: [NB][1024][CPAD] f16 NHWC    wt: [9][NT*16][CPAD] f16
//   Per-tap GEMM accumulation: M = pixels, N = cout, K = cin (32 per step).
//   Workgroup = 8 wave32; wave w owns output row (rg*8+w), both 16-px halves.
//   LDS tile [10 rows][34 cols][CPAD] f16 with halo + zero padding.
//
// Loop nest is deliberately: tap (NOT unrolled) -> kc -> n, with both M-tiles
// (x 0-15 and 16-31) consumed inside: each B fragment is loaded once and fed
// straight into two independent WMMAs. This keeps the live set ~100 VGPRs and
// prevents the compiler from hoisting all 72 B fragments (which previously
// spilled to scratch).
//
// A-fragment layout (16-bit A 16x32): lane half h = lane>=16,
//   element e -> K = e + h*16 (same order as B)  => two contiguous 8-half loads.
// B-fragment (32x16): lane n = lane&15, K = e + h*16 => one 16-half load.
// D (f32 16x16): vgpr r -> M = r + h*8, N = lane&15.
// ---------------------------------------------------------------------------
template <int CPAD, int NT>
__global__ __launch_bounds__(256) void conv3x3_wmma(
    const _Float16* __restrict__ act,
    const _Float16* __restrict__ wt,
    const float*    __restrict__ bias,
    _Float16*       __restrict__ out,
    int CpadOut, int doRelu)
{
    constexpr int COUT = NT * 16;
    constexpr int KCH  = CPAD / 32;
    constexpr int CH8  = CPAD / 8;
    __shared__ __align__(16) _Float16 tile[10 * 34 * CPAD];

    const int b   = blockIdx.x;   // block 0..383
    const int rg  = blockIdx.y;   // row group 0..3 (8 output rows each)
    const int tid = threadIdx.x;

    // ---- warm L2/L0 with this layer's weights (global_prefetch_b8)
    for (int o = tid * 64; o < 9 * COUT * CPAD; o += 256 * 64)
        __builtin_prefetch(wt + o, 0, 1);

    // ---- stage input tile (rows rg*8-1 .. rg*8+8, cols -1..32, zero padded)
    for (int i = tid; i < 10 * 34 * CH8; i += 256) {
        int cc   = i % CH8;
        int rest = i / CH8;
        int col  = rest % 34;
        int row  = rest / 34;
        int y = rg * 8 + row - 1;
        int x = col - 1;
        v8h v = {};
        if ((unsigned)y < 32u && (unsigned)x < 32u)
            v = *(const v8h*)(act + ((size_t)b * 1024 + y * 32 + x) * CPAD + cc * 8);
        *(v8h*)(tile + ((size_t)row * 34 + col) * CPAD + cc * 8) = v;
    }
    __syncthreads();

    const int wave = tid >> 5;    // wave w -> local output row w
    const int lane = tid & 31;
    const int hf   = lane >> 4;   // lane half (0: lanes 0-15, 1: lanes 16-31)
    const int m    = lane & 15;

    v8f acc[2][NT] = {};          // [x-half][n-tile], 2*NT*8 VGPRs

    #pragma unroll 1
    for (int tap = 0; tap < 9; ++tap) {
        const int ky = tap / 3, kx = tap % 3;
        // A: lane m holds pixel (row = wave, x = mi*16 + m), tap-shifted
        const _Float16* src0 = tile + ((size_t)(wave + ky) * 34 + (m + kx)) * CPAD;
        const _Float16* src1 = src0 + (size_t)16 * CPAD;
        const _Float16* wrow = wt + (size_t)tap * COUT * CPAD;
        #pragma unroll
        for (int kc = 0; kc < KCH; ++kc) {
            const int cb = kc * 32 + hf * 16;
            v8h a0lo = *(const v8h*)(src0 + cb);
            v8h a0hi = *(const v8h*)(src0 + cb + 8);
            v8h a1lo = *(const v8h*)(src1 + cb);
            v8h a1hi = *(const v8h*)(src1 + cb + 8);
            v16h A0, A1;
            #pragma unroll
            for (int e = 0; e < 8; ++e) {
                A0[e] = a0lo[e]; A0[e + 8] = a0hi[e];
                A1[e] = a1lo[e]; A1[e + 8] = a1hi[e];
            }
            #pragma unroll
            for (int n = 0; n < NT; ++n) {
                const int co = n * 16 + m;    // B: lane holds column N=m
                v16h B = *(const v16h*)(wrow + (size_t)co * CPAD + kc * 32 + hf * 16);
                acc[0][n] = __builtin_amdgcn_wmma_f32_16x16x32_f16(
                    false, A0, false, B, (short)0, acc[0][n], false, false);
                acc[1][n] = __builtin_amdgcn_wmma_f32_16x16x32_f16(
                    false, A1, false, B, (short)0, acc[1][n], false, false);
            }
        }
    }

    // ---- store D + bias (+ReLU) as f16 NHWC
    #pragma unroll
    for (int mi = 0; mi < 2; ++mi) {
        #pragma unroll
        for (int n = 0; n < NT; ++n) {
            const int co = n * 16 + m;
            const float bv = bias[co];
            #pragma unroll
            for (int r = 0; r < 8; ++r) {
                const int px = mi * 16 + r + hf * 8;     // D: M = r + hf*8
                const int p  = (rg * 8 + wave) * 32 + px;
                float v = acc[mi][n][r] + bv;
                if (doRelu) v = fmaxf(v, 0.f);
                out[((size_t)b * 1024 + p) * CpadOut + co] = (_Float16)v;
            }
        }
    }
}

// ---------------------------------------------------------------------------
// Fused pixel-shuffle + clip + crop + branch-select + reassembly -> img f32
// out img: [3][1920][1280]; src activations: [NB][32][32][48] f16 (pre-shuffle)
// ---------------------------------------------------------------------------
__global__ void assemble_out(const _Float16* __restrict__ cx, const _Float16* __restrict__ lt,
                             const float* __restrict__ clsvec, float* __restrict__ out)
{
    int idx = blockIdx.x * blockDim.x + threadIdx.x;
    int total = 3 * 1920 * 1280;
    if (idx >= total) return;
    int X = idx % 1280;
    int Y = (idx / 1280) % 1920;
    int c = idx / (1280 * 1920);
    int bh = Y / 80, yy = Y % 80;
    int bw = X / 80, xx = X % 80;
    int b = bh * NW + bw;
    int ys = yy + SF * PADN;   // crop offset 24 in 128x128 shuffled space
    int xs = xx + SF * PADN;
    int hh = ys >> 2, r1 = ys & 3;
    int ww = xs >> 2, r2 = xs & 3;
    int ch = c * 16 + r1 * 4 + r2;   // pixel-shuffle channel mapping
    const _Float16* src = (clsvec[b] > 0.5f) ? cx : lt;
    float v = (float)src[((size_t)b * 1024 + hh * 32 + ww) * 48 + ch];
    out[idx] = fminf(fmaxf(v, 0.f), 1.f);
}

// ---------------------------------------------------------------------------
// Host-side orchestration
// ---------------------------------------------------------------------------
static inline unsigned nblk(long long n) { return (unsigned)((n + 255) / 256); }

extern "C" void kernel_launch(void* const* d_in, const int* in_sizes, int n_in,
                              void* d_out, int out_size, void* d_ws, size_t ws_size,
                              hipStream_t stream)
{
    (void)in_sizes; (void)n_in; (void)out_size; (void)ws_size;
    const float* x    = (const float*)d_in[0];
    const float* cw1  = (const float*)d_in[1];  const float* cb1  = (const float*)d_in[2];
    const float* cw2  = (const float*)d_in[3];  const float* cb2  = (const float*)d_in[4];
    const float* cw3  = (const float*)d_in[5];  const float* cb3  = (const float*)d_in[6];
    const float* cw4  = (const float*)d_in[7];  const float* cb4  = (const float*)d_in[8];
    const float* cw5  = (const float*)d_in[9];  const float* cb5  = (const float*)d_in[10];
    const float* ltw1 = (const float*)d_in[11]; const float* ltb1 = (const float*)d_in[12];
    const float* ltw2 = (const float*)d_in[13]; const float* ltb2 = (const float*)d_in[14];
    const float* cxw1 = (const float*)d_in[15]; const float* cxb1 = (const float*)d_in[16];
    const float* cxw2 = (const float*)d_in[17]; const float* cxb2 = (const float*)d_in[18];
    const float* cxw3 = (const float*)d_in[19]; const float* cxb3 = (const float*)d_in[20];

    char* ws = (char*)d_ws;
    // Workspace layout with reuse (~145 MB peak):
    //   REG_CLS (39.3MB): h1 | h2a  ->  h2,h3 | h4a,h4  ->  lt2 output
    //   REG_X1  (50.3MB): cx1 out   ->  cx3 output
    //   REG_X2  (50.3MB): cx2 out   ->  lt1 output
    size_t off = 0;
    const size_t REG_CLS = off; off += 39321600;                 // 2x 19,660,800 B
    const size_t REG_X1  = off; off += 50331648;                 // 384*1024*64*2
    const size_t REG_X2  = off; off += 50331648;
    const size_t REG_BLK = off; off += (size_t)NB * 3 * 1024 * 4;
    const size_t OFF_LOG = off; off += 1536;
    const size_t OFF_CLS = off; off += 1536;
    const size_t OFF_WT2 = off; off += 9 * 64 * 64 * 2;          // cx2 weights f16
    const size_t OFF_WT3 = off; off += 9 * 48 * 64 * 2;          // cx3 weights f16
    const size_t OFF_WTL = off; off += 9 * 48 * 32 * 2;          // lt2 weights f16

    float* h1     = (float*)(ws + REG_CLS);
    float* h2a    = (float*)(ws + REG_CLS + 19660800);
    float* h2     = (float*)(ws + REG_CLS);                      // reuse h1 (dead)
    float* h3     = (float*)(ws + REG_CLS + 4915200);
    float* h4a    = (float*)(ws + REG_CLS + 19660800);           // reuse h2a (dead)
    float* h4     = (float*)(ws + REG_CLS + 19660800 + 1228800);
    float* logits = (float*)(ws + OFF_LOG);
    float* clsvec = (float*)(ws + OFF_CLS);
    float* blocks = (float*)(ws + REG_BLK);
    _Float16* cx1o = (_Float16*)(ws + REG_X1);
    _Float16* cx2o = (_Float16*)(ws + REG_X2);
    _Float16* cx3o = (_Float16*)(ws + REG_X1);                   // reuse cx1o (dead)
    _Float16* lt1o = (_Float16*)(ws + REG_X2);                   // reuse cx2o (dead)
    _Float16* lt2o = (_Float16*)(ws + REG_CLS);                  // reuse cls area (dead)
    _Float16* wt2  = (_Float16*)(ws + OFF_WT2);
    _Float16* wt3  = (_Float16*)(ws + OFF_WT3);
    _Float16* wtl  = (_Float16*)(ws + OFF_WTL);

    // ---- classifier branch (direct f32) ----
    conv2d_f32<<<nblk(32LL*480*320), 256, 0, stream>>>(x,  cw1, cb1, h1,  3, 32, 480, 320, 480, 320, 3, 1, 1, 1, 1);
    conv2d_f32<<<nblk(32LL*480*320), 256, 0, stream>>>(h1, cw2, cb2, h2a, 32, 32, 480, 320, 480, 320, 3, 1, 1, 1, 1);
    maxpool2_f32<<<nblk(32LL*240*160), 256, 0, stream>>>(h2a, h2, 32, 480, 320);
    conv2d_f32<<<nblk(16LL*240*160), 256, 0, stream>>>(h2, cw3, cb3, h3,  32, 16, 240, 160, 240, 160, 3, 1, 1, 1, 1);
    conv2d_f32<<<nblk(8LL*240*160), 256, 0, stream>>>(h3, cw4, cb4, h4a, 16, 8, 240, 160, 240, 160, 3, 1, 1, 1, 1);
    maxpool2_f32<<<nblk(8LL*120*80), 256, 0, stream>>>(h4a, h4, 8, 240, 160);
    conv2d_f32<<<nblk(24LL*16), 256, 0, stream>>>(h4, cw5, cb5, logits, 8, 1, 120, 80, 24, 16, 5, 5, 0, 0, 0);
    sigmoid384<<<2, 256, 0, stream>>>(logits, clsvec, (float*)d_out + 3LL * 1920 * 1280);

    // ---- block extraction + weight prep ----
    extract_blocks<<<nblk((long long)NB * 3 * 1024), 256, 0, stream>>>(x, blocks);
    prep_wt<<<nblk(9LL * 64 * 64), 256, 0, stream>>>(cxw2, wt2, 64, 64, 64);
    prep_wt<<<nblk(9LL * 48 * 64), 256, 0, stream>>>(cxw3, wt3, 48, 64, 64);
    prep_wt<<<nblk(9LL * 48 * 32), 256, 0, stream>>>(ltw2, wtl, 48, 16, 32);

    // ---- complex branch: cx1 direct, cx2/cx3 WMMA ----
    conv_block_c3<<<nblk((long long)NB * 1024 * 64), 256, 0, stream>>>(blocks, cxw1, cxb1, cx1o, 64, 64, 1);
    conv3x3_wmma<64, 4><<<dim3(NB, 4), 256, 0, stream>>>(cx1o, wt2, cxb2, cx2o, 64, 1);
    conv3x3_wmma<64, 3><<<dim3(NB, 4), 256, 0, stream>>>(cx2o, wt3, cxb3, cx3o, 48, 0);

    // ---- lightweight branch: lt1 direct, lt2 WMMA ----
    conv_block_c3<<<nblk((long long)NB * 1024 * 32), 256, 0, stream>>>(blocks, ltw1, ltb1, lt1o, 16, 32, 1);
    conv3x3_wmma<32, 3><<<dim3(NB, 4), 256, 0, stream>>>(lt1o, wtl, ltb2, lt2o, 48, 0);

    // ---- fused pixel-shuffle/clip/crop/select/reassemble ----
    assemble_out<<<nblk(3LL * 1920 * 1280), 256, 0, stream>>>(cx3o, lt2o, clsvec, (float*)d_out);
}
